// HierarchicalSequenceFusion_86346022518960
// MI455X (gfx1250) — compile-verified, matrix operand order checked by reading, not run on hardware
//
#include <hip/hip_runtime.h>

// ---------------------------------------------------------------------------
// Types for CDNA5 WMMA (wave32): bf16 A/B fragments (16 bf16 = 8 VGPRs),
// f32 C/D accumulator (8 VGPRs).
// ---------------------------------------------------------------------------
typedef __attribute__((ext_vector_type(16))) __bf16 v16bf;
typedef __attribute__((ext_vector_type(8)))  float  v8f;

union FragU {
    uint4 u[2];   // two 16B LDS loads (ds_load_b128)
    v16bf v;      // WMMA operand view
};

// ---------------------------------------------------------------------------
// Helpers
// ---------------------------------------------------------------------------
__device__ __forceinline__ unsigned int f32_to_bf16(float f) {
    unsigned int u = __float_as_uint(f);
    unsigned int r = u + 0x7FFFu + ((u >> 16) & 1u);   // round-to-nearest-even
    return r >> 16;
}

__device__ __forceinline__ float dot4(float4 a, float4 b) {
    return a.x * b.x + a.y * b.y + a.z * b.z + a.w * b.w;
}

__device__ __forceinline__ float4 f4add(float4 a, float4 b) {
    return make_float4(a.x + b.x, a.y + b.y, a.z + b.z, a.w + b.w);
}

#define HSF_H   1024
#define HSF_L   4096
#define HSF_B   8
#define HSF_ROWS (HSF_B * HSF_L)        // 32768
#define HSF_K4   (HSF_H / 4)

// ---------------------------------------------------------------------------
// 1) f32 -> bf16 row-major conversion (also used for agg before GEMM2)
// ---------------------------------------------------------------------------
__global__ __launch_bounds__(256) void hsf_convert_bf16(
    const float* __restrict__ in, unsigned int* __restrict__ out, int n4)
{
    int i = blockIdx.x * 256 + threadIdx.x;
    if (i >= n4) return;
    float4 v = ((const float4*)in)[i];
    unsigned int lo = f32_to_bf16(v.x) | (f32_to_bf16(v.y) << 16);
    unsigned int hi = f32_to_bf16(v.z) | (f32_to_bf16(v.w) << 16);
    ((uint2*)out)[i] = make_uint2(lo, hi);   // 2 packed bf16 per dword
}

// ---------------------------------------------------------------------------
// 2) Pack weight W[k][n] (f32, 1024x1024) -> Wp[n][kp] dword = {W[2kp][n], W[2kp+1][n]}
//    i.e. B^T with K-pairs packed, matching the A-operand VGPR layout.
// ---------------------------------------------------------------------------
__global__ __launch_bounds__(256) void hsf_pack_w(
    const float* __restrict__ w, unsigned int* __restrict__ wp)
{
    int i  = blockIdx.x * 256 + threadIdx.x;     // over 1024*512
    int n  = i >> 9;
    int kp = i & 511;
    float a = w[(size_t)(2 * kp)     * HSF_H + n];
    float b = w[(size_t)(2 * kp + 1) * HSF_H + n];
    wp[(size_t)n * 512 + kp] = f32_to_bf16(a) | (f32_to_bf16(b) << 16);
}

// ---------------------------------------------------------------------------
// 3) bf16 WMMA GEMM:  out[M x 1024] = A[M x 1024] * W + bias
//    Block: 256 thr (8 waves), tile 128x128, wave tile 32x64, K-step 32.
//    LDS rows padded to 80B (20-dword stride -> conflict-free b128 frag loads).
// ---------------------------------------------------------------------------
__global__ __launch_bounds__(256) void hsf_gemm_bias(
    const unsigned short* __restrict__ A,   // bf16 row-major, pitch 1024
    const unsigned int*   __restrict__ Wp,  // packed B^T, pitch 512 dwords
    const float*          __restrict__ bias,
    float*                __restrict__ out)
{
    __shared__ uint4 ldsA[128 * 5];   // 128 rows x 80B
    __shared__ uint4 ldsB[128 * 5];

    const int tid   = threadIdx.x;
    const int lane  = tid & 31;
    const int wid   = tid >> 5;
    const int mw    = wid & 3;        // 4 waves along M
    const int nw    = wid >> 2;       // 2 waves along N
    const int gm0   = blockIdx.x * 128;
    const int gn0   = blockIdx.y * 128;
    const int lrow  = lane & 15;
    const int lhalf = lane >> 4;

    const char* Ab = (const char*)A  + (size_t)gm0 * 2048;  // 1024 bf16/row
    const char* Bb = (const char*)Wp + (size_t)gn0 * 2048;  // 512 dwords/row

    v8f acc[2][4];
#pragma unroll
    for (int s = 0; s < 2; ++s)
#pragma unroll
        for (int t = 0; t < 4; ++t)
            acc[s][t] = (v8f){0.f, 0.f, 0.f, 0.f, 0.f, 0.f, 0.f, 0.f};

    for (int kk = 0; kk < 32; ++kk) {
        const int kb = kk * 64;       // byte offset of this K-slab within a row
#pragma unroll
        for (int it = 0; it < 2; ++it) {
            int c  = tid + it * 256;          // 512 chunks of 16B per tile
            int r  = c >> 2;
            int cx = c & 3;
            ldsA[r * 5 + cx] = *(const uint4*)(Ab + (size_t)r * 2048 + kb + cx * 16);
            ldsB[r * 5 + cx] = *(const uint4*)(Bb + (size_t)r * 2048 + kb + cx * 16);
        }
        __syncthreads();

        FragU afr[2], bfr[4];
#pragma unroll
        for (int s = 0; s < 2; ++s) {
            int row = mw * 32 + s * 16 + lrow;
            afr[s].u[0] = ldsA[row * 5 + lhalf];
            afr[s].u[1] = ldsA[row * 5 + lhalf + 2];
        }
#pragma unroll
        for (int t = 0; t < 4; ++t) {
            int row = nw * 64 + t * 16 + lrow;
            bfr[t].u[0] = ldsB[row * 5 + lhalf];
            bfr[t].u[1] = ldsB[row * 5 + lhalf + 2];
        }
#pragma unroll
        for (int s = 0; s < 2; ++s)
#pragma unroll
            for (int t = 0; t < 4; ++t)
                acc[s][t] = __builtin_amdgcn_wmma_f32_16x16x32_bf16(
                    false, afr[s].v, false, bfr[t].v,
                    (short)0, acc[s][t], false, false);
        __syncthreads();
    }

    // Epilogue: D layout -> m = vgpr + (lane>>4)*8, n = lane&15 within 16x16 tile
#pragma unroll
    for (int s = 0; s < 2; ++s) {
        int gmb = gm0 + mw * 32 + s * 16;
#pragma unroll
        for (int t = 0; t < 4; ++t) {
            int gn = gn0 + nw * 64 + t * 16 + lrow;
            float bv = bias[gn];
#pragma unroll
            for (int i = 0; i < 8; ++i) {
                int gm = gmb + i + lhalf * 8;
                out[(size_t)gm * HSF_H + gn] = acc[s][t][i] + bv;
            }
        }
    }
}

// ---------------------------------------------------------------------------
// 4) LayerNorm over H=1024, one wave per row; optional residual add.
// ---------------------------------------------------------------------------
__global__ __launch_bounds__(256) void hsf_layernorm(
    const float* __restrict__ x, const float* __restrict__ g,
    const float* __restrict__ b, const float* __restrict__ residual,
    float* __restrict__ out)
{
    int row  = blockIdx.x * 8 + (threadIdx.x >> 5);
    int lane = threadIdx.x & 31;
    if (row >= HSF_ROWS) return;

    const float4* xr = (const float4*)(x + (size_t)row * HSF_H);
    float4 v[8];
    float s = 0.f;
#pragma unroll
    for (int j = 0; j < 8; ++j) {
        v[j] = xr[lane + j * 32];
        s += v[j].x + v[j].y + v[j].z + v[j].w;
    }
#pragma unroll
    for (int m = 16; m >= 1; m >>= 1) s += __shfl_xor(s, m, 32);
    float mu = s * (1.0f / HSF_H);

    float var = 0.f;
#pragma unroll
    for (int j = 0; j < 8; ++j) {
        float dx = v[j].x - mu, dy = v[j].y - mu, dz = v[j].z - mu, dw = v[j].w - mu;
        var += dx * dx + dy * dy + dz * dz + dw * dw;
    }
#pragma unroll
    for (int m = 16; m >= 1; m >>= 1) var += __shfl_xor(var, m, 32);
    float rstd = rsqrtf(var * (1.0f / HSF_H) + 1e-5f);

    const float4* g4 = (const float4*)g;
    const float4* b4 = (const float4*)b;
    const float4* r4 = residual ? (const float4*)(residual + (size_t)row * HSF_H) : nullptr;
    float4* o4 = (float4*)(out + (size_t)row * HSF_H);
#pragma unroll
    for (int j = 0; j < 8; ++j) {
        int h = lane + j * 32;
        float4 gg = g4[h], bb = b4[h];
        float4 o;
        o.x = (v[j].x - mu) * rstd * gg.x + bb.x;
        o.y = (v[j].y - mu) * rstd * gg.y + bb.y;
        o.z = (v[j].z - mu) * rstd * gg.z + bb.z;
        o.w = (v[j].w - mu) * rstd * gg.w + bb.w;
        if (r4) { float4 rr = r4[h]; o.x += rr.x; o.y += rr.y; o.z += rr.z; o.w += rr.w; }
        o4[h] = o;
    }
}

// ---------------------------------------------------------------------------
// 5) Hierarchical fusion: one wave per row; level embeddings (4x1024) in LDS.
// ---------------------------------------------------------------------------
__global__ __launch_bounds__(256) void hsf_fusion(
    const float* __restrict__ t, const float* __restrict__ masks,
    const float* __restrict__ emb, float* __restrict__ out)
{
    __shared__ float4 semb[4 * HSF_K4];   // 16KB
    for (int i = threadIdx.x; i < 4 * HSF_K4; i += 256)
        semb[i] = ((const float4*)emb)[i];
    __syncthreads();

    int row  = blockIdx.x * 8 + (threadIdx.x >> 5);
    int lane = threadIdx.x & 31;
    if (row >= HSF_ROWS) return;

    const float4* tr = (const float4*)(t + (size_t)row * HSF_H);
    float4 tv[8];
    float s0 = 0.f, s1 = 0.f, s2 = 0.f, s3 = 0.f;
#pragma unroll
    for (int j = 0; j < 8; ++j) {
        int h = lane + j * 32;
        tv[j] = tr[h];
        s0 += dot4(tv[j], semb[h]);
        s1 += dot4(tv[j], semb[HSF_K4 + h]);
        s2 += dot4(tv[j], semb[2 * HSF_K4 + h]);
        s3 += dot4(tv[j], semb[3 * HSF_K4 + h]);
    }
#pragma unroll
    for (int m = 16; m >= 1; m >>= 1) {
        s0 += __shfl_xor(s0, m, 32);
        s1 += __shfl_xor(s1, m, 32);
        s2 += __shfl_xor(s2, m, 32);
        s3 += __shfl_xor(s3, m, 32);
    }

    float4 mk = ((const float4*)masks)[row];
    float w0 = mk.x > 0.5f ? expf(s0) : 0.f;
    float w1 = mk.y > 0.5f ? expf(s1) : 0.f;
    float w2 = mk.z > 0.5f ? expf(s2) : 0.f;
    float w3 = mk.w > 0.5f ? expf(s3) : 0.f;
    float tot = w0 + w1 + w2 + w3;
    float inv = (tot > 1e-8f) ? (1.f / tot) : 1.f;
    w0 *= inv; w1 *= inv; w2 *= inv; w3 *= inv;

    float4 fv[8];
    float dg = 0.f;
#pragma unroll
    for (int j = 0; j < 8; ++j) {
        int h = lane + j * 32;
        float4 e0 = semb[h], e1 = semb[HSF_K4 + h];
        float4 e2 = semb[2 * HSF_K4 + h], e3 = semb[3 * HSF_K4 + h];
        float4 f;
        f.x = w0 * e0.x + w1 * e1.x + w2 * e2.x + w3 * e3.x;
        f.y = w0 * e0.y + w1 * e1.y + w2 * e2.y + w3 * e3.y;
        f.z = w0 * e0.z + w1 * e1.z + w2 * e2.z + w3 * e3.z;
        f.w = w0 * e0.w + w1 * e1.w + w2 * e2.w + w3 * e3.w;
        fv[j] = f;
        dg += dot4(f, tv[j]);
    }
#pragma unroll
    for (int m = 16; m >= 1; m >>= 1) dg += __shfl_xor(dg, m, 32);
    float gate = 1.f / (1.f + expf(-dg * (1.0f / HSF_H)));
    float og   = 1.f - gate;

    float4* o4 = (float4*)(out + (size_t)row * HSF_H);
#pragma unroll
    for (int j = 0; j < 8; ++j) {
        int h = lane + j * 32;
        float4 o;
        o.x = gate * fv[j].x + og * tv[j].x;
        o.y = gate * fv[j].y + og * tv[j].y;
        o.z = gate * fv[j].z + og * tv[j].z;
        o.w = gate * fv[j].w + og * tv[j].w;
        o4[h] = o;
    }
}

// ---------------------------------------------------------------------------
// 6) Multi-scale windowed means (scales 1,3,7,15) with LDS tile + +/-7 halo.
//    Block tile: 64 L x 64 H (16 float4 columns).
// ---------------------------------------------------------------------------
__global__ __launch_bounds__(256) void hsf_mscale(
    const float* __restrict__ fused, float* __restrict__ agg)
{
    __shared__ float4 tile[78 * 16];
    const int L  = HSF_L;
    const int bb = blockIdx.z;
    const int l0 = blockIdx.x * 64;
    const int c0 = blockIdx.y * 16;
    const float4* src = (const float4*)fused + (size_t)bb * L * HSF_K4;

    for (int i = threadIdx.x; i < 78 * 16; i += 256) {
        int r = i >> 4, c = i & 15;
        int gl = l0 - 7 + r;
        float4 v = make_float4(0.f, 0.f, 0.f, 0.f);
        if (gl >= 0 && gl < L) v = src[(size_t)gl * HSF_K4 + c0 + c];
        tile[i] = v;
    }
    __syncthreads();

    const int c  = threadIdx.x & 15;
    const int lb = threadIdx.x >> 4;
    const float w3  = 1.f / (1.f + 1.09861229f);
    const float w7  = 1.f / (1.f + 1.94591015f);
    const float w15 = 1.f / (1.f + 2.70805020f);

    float4* dst = (float4*)agg + (size_t)bb * L * HSF_K4;
#pragma unroll
    for (int it = 0; it < 4; ++it) {
        int l  = lb + it * 16;
        int gl = l0 + l;
        float4 z = make_float4(0.f, 0.f, 0.f, 0.f);
        float4 s1 = z, s3 = z, s7 = z, s15 = z;
#pragma unroll
        for (int d = -7; d <= 7; ++d) {
            int g = gl + d;
            if (g < 0 || g >= L) continue;
            float4 v = tile[(l + 7 + d) * 16 + c];
            s15 = f4add(s15, v);
            if (d >= -3 && d <= 3) s7 = f4add(s7, v);
            if (d >= -1 && d <= 1) s3 = f4add(s3, v);
            if (d == 0) s1 = v;
        }
        float i3  = w3  / (float)(min(gl + 1, L - 1) - max(gl - 1, 0) + 1);
        float i7  = w7  / (float)(min(gl + 3, L - 1) - max(gl - 3, 0) + 1);
        float i15 = w15 / (float)(min(gl + 7, L - 1) - max(gl - 7, 0) + 1);
        float4 o;
        o.x = s1.x + s3.x * i3 + s7.x * i7 + s15.x * i15;
        o.y = s1.y + s3.y * i3 + s7.y * i7 + s15.y * i15;
        o.z = s1.z + s3.z * i3 + s7.z * i7 + s15.z * i15;
        o.w = s1.w + s3.w * i3 + s7.w * i7 + s15.w * i15;
        dst[(size_t)gl * HSF_K4 + c0 + c] = o;
    }
}

// ---------------------------------------------------------------------------
// Launcher
// ---------------------------------------------------------------------------
extern "C" void kernel_launch(void* const* d_in, const int* in_sizes, int n_in,
                              void* d_out, int out_size, void* d_ws, size_t ws_size,
                              hipStream_t stream) {
    (void)in_sizes; (void)n_in; (void)out_size; (void)ws_size;

    const float* seq   = (const float*)d_in[0];
    const float* masks = (const float*)d_in[1];
    const float* emb   = (const float*)d_in[2];
    const float* lt_w  = (const float*)d_in[3];
    const float* lt_b  = (const float*)d_in[4];
    const float* ln1g  = (const float*)d_in[5];
    const float* ln1b  = (const float*)d_in[6];
    const float* ft_w  = (const float*)d_in[7];
    const float* ft_b  = (const float*)d_in[8];
    const float* ln2g  = (const float*)d_in[9];
    const float* ln2b  = (const float*)d_in[10];
    float* out = (float*)d_out;

    char* ws = (char*)d_ws;
    const size_t N_ELEM = (size_t)HSF_ROWS * HSF_H;        // 33,554,432
    unsigned int*   seqb = (unsigned int*)(ws);            // bf16 acts (64MB), reused for agg_bf16
    unsigned int*   ltp  = (unsigned int*)(ws + 67108864);
    unsigned int*   ftp  = (unsigned int*)(ws + 69206016);
    float*          bufA = (float*)(ws + 71303168);        // y1 -> agg
    float*          bufB = (float*)(ws + 205520896);       // t  -> y2
    float*          bufC = (float*)(ws + 339738624);       // fused

    const int n4 = (int)(N_ELEM / 4);

    // Stage 0: precision conversion / weight packing for the WMMA path
    hsf_convert_bf16<<<n4 / 256, 256, 0, stream>>>(seq, seqb, n4);
    hsf_pack_w<<<(1024 * 512) / 256, 256, 0, stream>>>(lt_w, ltp);
    hsf_pack_w<<<(1024 * 512) / 256, 256, 0, stream>>>(ft_w, ftp);

    // Stage 1: t = LN(seq @ lt_w + lt_b)
    hsf_gemm_bias<<<dim3(HSF_ROWS / 128, HSF_H / 128), 256, 0, stream>>>(
        (const unsigned short*)seqb, ltp, lt_b, bufA);
    hsf_layernorm<<<HSF_ROWS / 8, 256, 0, stream>>>(bufA, ln1g, ln1b, nullptr, bufB);

    // Stage 2: hierarchical fusion
    hsf_fusion<<<HSF_ROWS / 8, 256, 0, stream>>>(bufB, masks, emb, bufC);

    // Stage 3: multi-scale aggregation
    hsf_mscale<<<dim3(HSF_L / 64, HSF_H / 64, HSF_B), 256, 0, stream>>>(bufC, bufA);

    // Stage 4: out = LN(agg @ ft_w + ft_b) + seq
    hsf_convert_bf16<<<n4 / 256, 256, 0, stream>>>(bufA, seqb, n4);
    hsf_gemm_bias<<<dim3(HSF_ROWS / 128, HSF_H / 128), 256, 0, stream>>>(
        (const unsigned short*)seqb, ftp, ft_b, bufB);
    hsf_layernorm<<<HSF_ROWS / 8, 256, 0, stream>>>(bufB, ln2g, ln2b, seq, out);
}
